// ByteNetDecoder_3281355014270
// MI455X (gfx1250) — matrix operand order repaired and checked by reading, not compiled, not run on hardware
//
#include <hip/hip_runtime.h>
#include <hip/hip_bf16.h>

#define ENC_DIMX 256
#define C1 128
#define C2X 256
#define NLAYERS 5
#define MAX_TT 48
#define T_ENCX 64
#define BATCHX 8
#define NUM_LAB 64
#define STOP_TOK 6
#define START_TOK 5
#define NTHREADS 512
#define NWAVES 16
#define FRAG 512                 // bf16 elems per (ktile): 32 lanes x 16
#define RING_SLOTS (MAX_TT + 1)  // slot MAX_TT = constant PAD-state fragment

typedef __attribute__((ext_vector_type(16))) __bf16 v16bf;
typedef __attribute__((ext_vector_type(8)))  float  v8f;

__device__ __forceinline__ unsigned short f2bf(float f) {
  unsigned int u = __float_as_uint(f);
  u += 0x7FFFu + ((u >> 16) & 1u);          // round-to-nearest-even
  return (unsigned short)(u >> 16);
}

union FR { v16bf v; uint4 q[2]; };

// 16-bit A-matrix 16x32 layout: lanes 0-15 hold K in {0..7,16..23}, lanes 16-31 hold {8..15,24..31}
__device__ __forceinline__ int a_kloc(int e, int half) {
  return (e < 8) ? (8 * half + e) : (8 * half + e + 8);
}

// cooperative: pack A fragments (ktiles x 512 bf16) from a f32 [16 x stride] buffer
__device__ __forceinline__ void packA(const float* __restrict__ src, int stride, int relu,
                                      unsigned short* __restrict__ dst, int ktiles) {
  for (int idx = threadIdx.x; idx < ktiles * FRAG; idx += NTHREADS) {
    int e = idx & 15, lane = (idx >> 4) & 31, kt = idx >> 9;
    int k = kt * 32 + a_kloc(e, lane >> 4);
    float v = src[(lane & 15) * stride + k];
    if (relu) v = fmaxf(v, 0.0f);
    dst[idx] = f2bf(v);
  }
}

// ---- per-ktile A-fragment base selectors (fragments are pre-packed, loads contiguous) ----
struct FragLin {
  const unsigned short* base;
  __device__ const unsigned short* operator()(int kt) const { return base + (size_t)kt * FRAG; }
};
struct FragConv {  // K=384 as 3 taps of 128 (4 ktiles each): h1 at t-2d, t-d, t
  const unsigned short* cur;    // LDS staging of slot t
  const unsigned short* ringL;  // this layer's ring (RING_SLOTS x 4 x FRAG)
  int t, d;
  __device__ const unsigned short* operator()(int kt) const {
    int tap = kt >> 2, ktl = kt & 3;
    if (tap == 2) return cur + ktl * FRAG;
    int tsrc = t - (2 - tap) * d;
    int slot = (tsrc < 0) ? MAX_TT : tsrc;   // negative time -> constant PAD-state slot
    return ringL + ((size_t)slot * 4 + ktl) * FRAG;
  }
};

// ---- (16 x 32K) @ (32K x 16N) GEMM via V_WMMA_F32_16X16X32_BF16; N-tiles across 16 waves ----
template <typename FB>
__device__ __forceinline__ void gemm(const FB& fb, const unsigned short* __restrict__ Bpack,
                                     int ktiles, int ntiles, float* __restrict__ out, int ostride,
                                     const float* __restrict__ bias, int reluOut, int accum) {
  const int lane = threadIdx.x & 31;
  const int wave = threadIdx.x >> 5;
  for (int nt = wave; nt < ntiles; nt += NWAVES) {
    v8f acc = {};
    for (int kt = 0; kt < ktiles; ++kt) {
      FR a, b;
      const uint4* ap = (const uint4*)(fb(kt) + (size_t)lane * 16);
      a.q[0] = ap[0]; a.q[1] = ap[1];
      const uint4* bp = (const uint4*)(Bpack + (((size_t)(nt * ktiles + kt)) * 32 + lane) * 16);
      b.q[0] = bp[0]; b.q[1] = bp[1];
      acc = __builtin_amdgcn_wmma_f32_16x16x32_bf16(false, a.v, false, b.v, (short)0, acc,
                                                    false, false);
    }
    int half = lane >> 4;
    int n = nt * 16 + (lane & 15);
    float bv = bias ? bias[n] : 0.0f;
#pragma unroll
    for (int v = 0; v < 8; ++v) {
      int m = v + 8 * half;                 // C/D layout: lanes 0-15 -> M=v, lanes 16-31 -> M=v+8
      float r = acc[v] + bv;
      if (reluOut) r = fmaxf(r, 0.0f);
      float* p = out + m * ostride + n;
      if (accum) *p += r; else *p = r;
    }
  }
}

// ---- prologue: pack f32 (K,N) weights into fragment-linear bf16 B tiles ----
// B 32x16 layout: lane = column (n = lane&15), K matches A: k = kt*32 + a_kloc(e, lane>>4)
__global__ void pack_b_kernel(const float* __restrict__ W, unsigned short* __restrict__ out,
                              int Kdim, int Ndim, int srcStride, int colOff) {
  int idx = blockIdx.x * blockDim.x + threadIdx.x;
  if (idx >= Kdim * Ndim) return;
  int e = idx & 15;
  int lane = (idx >> 4) & 31;
  int tile = idx >> 9;
  int ktiles = Kdim >> 5;
  int nt = tile / ktiles, kt = tile - nt * ktiles;
  int n = nt * 16 + (lane & 15);
  int k = kt * 32 + a_kloc(e, lane >> 4);
  out[idx] = f2bf(W[k * srcStride + colOff + n]);
}

// ---- prologue: pre-pack encoder A-fragments for every step t ----
__global__ void pack_enca_kernel(const float* __restrict__ enc, unsigned short* __restrict__ out) {
  int idx = blockIdx.x * blockDim.x + threadIdx.x;
  if (idx >= MAX_TT * 8 * FRAG) return;
  int e = idx & 15, lane = (idx >> 4) & 31, kt = (idx >> 9) & 7, t = idx >> 12;
  int m = lane & 15;
  int k = kt * 32 + a_kloc(e, lane >> 4);
  float v = (m < BATCHX) ? enc[(m * ENC_DIMX + k) * T_ENCX + t] : 0.0f;
  out[idx] = f2bf(v);
}

// ---- prologue: embed_eff = embed @ w_in  (folds the 1x1 input conv into the table) ----
__global__ void embed_eff_kernel(const float* __restrict__ embed, const float* __restrict__ w_in,
                                 float* __restrict__ out) {
  int n = threadIdx.x, l = blockIdx.x;
  float s = 0.0f;
  for (int k = 0; k < C2X; ++k) s += embed[l * C2X + k] * w_in[k * C2X + n];
  out[l * C2X + n] = s;
}

// ---- main persistent decoder: one workgroup, 16 wave32, 48 sequential steps ----
__global__ __launch_bounds__(NTHREADS)
void decode_kernel(const float* __restrict__ embed_eff,
                   const float* __restrict__ b_in, const float* __restrict__ b_enc,
                   const float* __restrict__ res_w1, const float* __restrict__ res_b1,
                   const float* __restrict__ res_wd, const float* __restrict__ res_bd,
                   const float* __restrict__ res_w2, const float* __restrict__ res_b2,
                   const float* __restrict__ b_o1, const float* __restrict__ b_o2,
                   const unsigned short* __restrict__ w_enc_p, const unsigned short* __restrict__ w1_p,
                   const unsigned short* __restrict__ wd_p, const unsigned short* __restrict__ w2_p,
                   const unsigned short* __restrict__ o1_p, const unsigned short* __restrict__ o2_p,
                   const unsigned short* __restrict__ encA,
                   unsigned short* __restrict__ ringA, float* __restrict__ out) {
  __shared__ float s_x[16 * C2X];            // residual stream (16 x 256)
  __shared__ float s_h[16 * C2X];            // h1 (16x128) | h2 (16x128); reused as o1 half (16x256)
  __shared__ float s_logits[16 * NUM_LAB];
  __shared__ float s_h1pad[NLAYERS * C1];
  __shared__ float s_xpad[C2X];
  __shared__ float s_tmp[C2X];
  __shared__ unsigned short s_afragA[8 * FRAG];   // staged A fragments (max K=256)
  __shared__ unsigned short s_afragB[8 * FRAG];
  __shared__ int s_label[16];
  __shared__ int s_stop[BATCHX];

  const int tid = threadIdx.x;
  float* s_h1 = s_h;
  float* s_h2 = s_h + 16 * C1;

  if (tid < 16) s_label[tid] = START_TOK;
  if (tid < BATCHX) s_stop[tid] = MAX_TT;

  // ---- PAD-state: activations for positions g < 0 (constant input: PAD token, zero encoding) ----
  for (int n = tid; n < C2X; n += NTHREADS)
    s_xpad[n] = embed_eff[/*PAD=0*/ n] + b_in[n] + b_enc[n];
  __syncthreads();
  for (int i = 0; i < NLAYERS; ++i) {
    for (int n = tid; n < C2X; n += NTHREADS) s_tmp[n] = fmaxf(s_xpad[n], 0.0f);
    __syncthreads();
    for (int c = tid; c < C1; c += NTHREADS) {
      float s = res_b1[i * C1 + c];
      for (int k = 0; k < C2X; ++k) s += s_tmp[k] * res_w1[(i * C2X + k) * C1 + c];
      s_h1pad[i * C1 + c] = fmaxf(s, 0.0f);
    }
    __syncthreads();
    for (int c = tid; c < C1; c += NTHREADS) {        // all 3 taps see the same constant h1_pad
      float s = res_bd[i * C1 + c];
      for (int k = 0; k < 3 * C1; ++k)
        s += s_h1pad[i * C1 + (k & (C1 - 1))] * res_wd[(i * 3 * C1 + k) * C1 + c];
      s_h1[c] = fmaxf(s, 0.0f);                       // h2_pad temp
    }
    __syncthreads();
    for (int n = tid; n < C2X; n += NTHREADS) {
      float s = res_b2[i * C2X + n];
      for (int c = 0; c < C1; ++c) s += s_h1[c] * res_w2[(i * C1 + c) * C2X + n];
      s_xpad[n] += s;
    }
    __syncthreads();
  }

  // pre-pack PAD-state h1 fragments into ring slot MAX_TT for every layer
  for (int idx = tid; idx < NLAYERS * 4 * FRAG; idx += NTHREADS) {
    int i = idx / (4 * FRAG);
    int r = idx & (4 * FRAG - 1);
    int e = r & 15, lane = (r >> 4) & 31, ktl = r >> 9;
    int k = ktl * 32 + a_kloc(e, lane >> 4);
    ringA[((size_t)i * RING_SLOTS + MAX_TT) * 4 * FRAG + r] = f2bf(s_h1pad[i * C1 + k]);
  }
  __threadfence();
  __syncthreads();

  const int dil[NLAYERS] = {1, 2, 4, 8, 16};

  // ---- sequential decode ----
  for (int t = 0; t < MAX_TT; ++t) {
    // x = embed_eff[label] + b_in + b_enc  (pad rows zero)
    for (int idx = tid; idx < 16 * C2X; idx += NTHREADS) {
      int m = idx >> 8, n = idx & (C2X - 1);
      s_x[idx] = (m < BATCHX) ? (embed_eff[s_label[m] * C2X + n] + b_in[n] + b_enc[n]) : 0.0f;
    }
    __syncthreads();
    // x += enc(t) @ w_enc   (enc A-fragments pre-packed in prologue)
    gemm(FragLin{encA + (size_t)t * 8 * FRAG}, w_enc_p, 8, 16, s_x, C2X, nullptr, 0, 1);
    __syncthreads();

    for (int i = 0; i < NLAYERS; ++i) {
      // h1 = relu(relu(x) @ w1 + b1)
      packA(s_x, C2X, 1, s_afragA, 8);
      __syncthreads();
      gemm(FragLin{s_afragA}, w1_p + (size_t)i * C2X * C1, 8, 8, s_h1, C1, res_b1 + i * C1, 1, 0);
      __syncthreads();
      // pack h1(t) fragments, publish to ring slot t for future steps' dilated taps
      packA(s_h1, C1, 0, s_afragA, 4);
      __syncthreads();
      {
        const unsigned int* sa = (const unsigned int*)s_afragA;
        unsigned int* rg = (unsigned int*)(ringA + ((size_t)i * RING_SLOTS + t) * 4 * FRAG);
        for (int idx = tid; idx < 4 * FRAG / 2; idx += NTHREADS) rg[idx] = sa[idx];
      }
      __threadfence();
      __syncthreads();
      // h2 = relu(sum_taps h1(t-(2-w)*d) @ wd[w] + bd)
      gemm(FragConv{s_afragA, ringA + (size_t)i * RING_SLOTS * 4 * FRAG, t, dil[i]},
           wd_p + (size_t)i * 3 * C1 * C1, 12, 8, s_h2, C1, res_bd + i * C1, 1, 0);
      __syncthreads();
      // x += h2 @ w2 + b2
      packA(s_h2, C1, 0, s_afragB, 4);
      __syncthreads();
      gemm(FragLin{s_afragB}, w2_p + (size_t)i * C1 * C2X, 4, 16, s_x, C2X, res_b2 + i * C2X, 0, 1);
      __syncthreads();
    }

    // head: logits = relu(relu(x) @ w_o1 + b_o1) @ w_o2 + b_o2, K of second GEMM split in 2 passes
    for (int idx = tid; idx < 16 * NUM_LAB; idx += NTHREADS)
      s_logits[idx] = b_o2[idx & (NUM_LAB - 1)];
    packA(s_x, C2X, 1, s_afragA, 8);
    __syncthreads();
    for (int p = 0; p < 2; ++p) {
      gemm(FragLin{s_afragA}, o1_p + (size_t)p * C2X * 256, 8, 16, s_h, 256, b_o1 + p * 256, 1, 0);
      __syncthreads();
      packA(s_h, 256, 0, s_afragB, 8);
      __syncthreads();
      gemm(FragLin{s_afragB}, o2_p + (size_t)p * 256 * NUM_LAB, 8, 4, s_logits, NUM_LAB,
           nullptr, 0, 1);
      __syncthreads();
    }

    // emit logits (B, NUM_LABELS, T) and argmax -> next token
    for (int idx = tid; idx < BATCHX * NUM_LAB; idx += NTHREADS) {
      int b = idx >> 6, n = idx & (NUM_LAB - 1);
      out[(b * NUM_LAB + n) * MAX_TT + t] = s_logits[b * NUM_LAB + n];
    }
    if (tid < BATCHX) {
      int b = tid;
      float best = s_logits[b * NUM_LAB];
      int bi = 0;
      for (int n = 1; n < NUM_LAB; ++n) {
        float v = s_logits[b * NUM_LAB + n];
        if (v > best) { best = v; bi = n; }
      }
      s_label[b] = bi;
      if (bi == STOP_TOK && t < s_stop[b]) s_stop[b] = t;   // first STOP step == length
    }
    __syncthreads();
  }

  if (tid < BATCHX)
    ((int*)(out + BATCHX * NUM_LAB * MAX_TT))[tid] = s_stop[tid];
}

extern "C" void kernel_launch(void* const* d_in, const int* in_sizes, int n_in,
                              void* d_out, int out_size, void* d_ws, size_t ws_size,
                              hipStream_t stream) {
  const float* enc_seq = (const float*)d_in[0];
  const float* embed   = (const float*)d_in[1];
  const float* w_in    = (const float*)d_in[2];
  const float* b_in    = (const float*)d_in[3];
  const float* w_enc   = (const float*)d_in[4];
  const float* b_enc   = (const float*)d_in[5];
  const float* res_w1  = (const float*)d_in[6];
  const float* res_b1  = (const float*)d_in[7];
  const float* res_wd  = (const float*)d_in[8];
  const float* res_bd  = (const float*)d_in[9];
  const float* res_w2  = (const float*)d_in[10];
  const float* res_b2  = (const float*)d_in[11];
  const float* w_o1    = (const float*)d_in[12];
  const float* b_o1    = (const float*)d_in[13];
  const float* w_o2    = (const float*)d_in[14];
  const float* b_o2    = (const float*)d_in[15];
  float* out = (float*)d_out;

  // workspace carve-up
  size_t off = 0;
  auto carve_us = [&](size_t elems) {
    unsigned short* p = (unsigned short*)((char*)d_ws + off);
    off += elems * sizeof(unsigned short);
    return p;
  };
  unsigned short* w_enc_p = carve_us((size_t)256 * 256);
  unsigned short* w1_p    = carve_us((size_t)5 * 256 * 128);
  unsigned short* wd_p    = carve_us((size_t)5 * 384 * 128);
  unsigned short* w2_p    = carve_us((size_t)5 * 128 * 256);
  unsigned short* o1_p    = carve_us((size_t)2 * 256 * 256);
  unsigned short* o2_p    = carve_us((size_t)2 * 256 * 64);
  unsigned short* encA    = carve_us((size_t)MAX_TT * 8 * FRAG);
  unsigned short* ringA   = carve_us((size_t)NLAYERS * RING_SLOTS * 4 * FRAG);
  off = (off + 255) & ~(size_t)255;
  float* embed_eff = (float*)((char*)d_ws + off); off += (size_t)64 * 256 * sizeof(float);

  auto pack = [&](const float* W, unsigned short* dst, int K, int N, int stride, int colOff) {
    int total = K * N;
    pack_b_kernel<<<(total + 255) / 256, 256, 0, stream>>>(W, dst, K, N, stride, colOff);
  };
  pack(w_enc, w_enc_p, 256, 256, 256, 0);
  for (int i = 0; i < 5; ++i) pack(res_w1 + (size_t)i * 256 * 128, w1_p + (size_t)i * 256 * 128, 256, 128, 128, 0);
  for (int i = 0; i < 5; ++i) pack(res_wd + (size_t)i * 384 * 128, wd_p + (size_t)i * 384 * 128, 384, 128, 128, 0);
  for (int i = 0; i < 5; ++i) pack(res_w2 + (size_t)i * 128 * 256, w2_p + (size_t)i * 128 * 256, 128, 256, 256, 0);
  for (int p = 0; p < 2; ++p) pack(w_o1, o1_p + (size_t)p * 256 * 256, 256, 256, 512, p * 256);
  for (int p = 0; p < 2; ++p) pack(w_o2 + (size_t)p * 256 * 64, o2_p + (size_t)p * 256 * 64, 256, 64, 64, 0);

  {
    int total = MAX_TT * 8 * FRAG;
    pack_enca_kernel<<<(total + 255) / 256, 256, 0, stream>>>(enc_seq, encA);
  }
  embed_eff_kernel<<<64, 256, 0, stream>>>(embed, w_in, embed_eff);

  decode_kernel<<<1, NTHREADS, 0, stream>>>(
      embed_eff, b_in, b_enc,
      res_w1, res_b1, res_wd, res_bd, res_w2, res_b2,
      b_o1, b_o2,
      w_enc_p, w1_p, wd_p, w2_p, o1_p, o2_p,
      encA, ringA, out);
}
